// TransformerBlock_9912784519803
// MI455X (gfx1250) — matrix operand order repaired and checked by reading, not compiled, not run on hardware
//
#include <hip/hip_runtime.h>
#include <math.h>

// ---------------------------------------------------------------------------
// MI455X (gfx1250, wave32) transformer block forward.
// bf16 matrix path with f32 accumulation via v_wmma_f32_16x16x32_bf16.
// Flash-attention softmax (no SxS materialization). LDS staging uses the
// CDNA5 async global->LDS copies (ASYNCcnt) when the builtin is available.
// ---------------------------------------------------------------------------

typedef __attribute__((ext_vector_type(16))) __bf16 v16bf;
typedef __attribute__((ext_vector_type(8)))  float  v8f;
typedef __attribute__((ext_vector_type(4)))  int    v4i;
typedef __bf16 bf16;

union Frag16 { v16bf v; uint4 q[2]; bf16 h[16]; };

#if defined(__has_builtin)
#if __has_builtin(__builtin_amdgcn_global_load_async_to_lds_b128)
#define USE_ASYNC_LDS 1
#endif
#endif

#ifdef USE_ASYNC_LDS
// Per CDNA5 ISA: copies global memory into LDS at matching offsets, tracked on ASYNCcnt.
// Builtin signature (from clang diagnostic): (v4i addrspace(1)*, v4i addrspace(3)*,
// imm offset, imm cpol).
#define ASYNC_LD16(ldsp, gp, OFF)                                           \
    __builtin_amdgcn_global_load_async_to_lds_b128(                         \
        (__attribute__((address_space(1))) v4i*)(void*)(gp),                \
        (__attribute__((address_space(3))) v4i*)(void*)(ldsp), (OFF), 0)
#define ASYNC_WAIT() asm volatile("s_wait_asynccnt 0x0" ::: "memory")
#endif

// WMMA fragment layout (CDNA5 ISA 7.12.2, wave32):
//  A (16x32 bf16): lane l -> row l%16; VGPR0-3 = K {0..7|8..15}, VGPR4-7 = +16.
//  B (32x16 bf16): lane l -> col l%16; lanes 0-15 K=0..7,16..23, lanes 16-31 K=8..15,24..31.
//  C/D (16x16 f32): VGPR r, lanes 0-15 -> (M=r, N=lane); lanes 16-31 -> (M=r+8).

// ---------------------------------------------------------------------------
// Generic GEMM: out[M,N] = epilogue( A[M,K](bf16) * W[N,K](bf16)^T )
// 256 thr (8 waves), tile 128x128, K-step 32; wave = 32(M) x 64(N) = 2x4 WMMA.
// ---------------------------------------------------------------------------
__global__ __launch_bounds__(256)
void gemm_bf16_wmma(const bf16* __restrict__ A, const bf16* __restrict__ Wt,
                    const float* __restrict__ bias, const float* __restrict__ residual,
                    const int* __restrict__ rowvalid,
                    bf16* __restrict__ outb, float* __restrict__ outf,
                    int M, int N, int K, int do_gelu)
{
    __shared__ bf16 As[128][40];   // 128 x 32, padded stride 40 (80B)
    __shared__ bf16 Bs[128][40];

    const int tid  = threadIdx.x;
    const int lane = tid & 31;
    const int w    = tid >> 5;
    const int wm   = w & 3;
    const int wn   = w >> 2;
    const int m0   = blockIdx.y * 128;
    const int n0   = blockIdx.x * 128;

    v8f acc[2][4] = {};

    const int lrow = tid >> 1;
    const int lseg = (tid & 1) * 16;

    for (int kk = 0; kk < K; kk += 32) {
#ifdef USE_ASYNC_LDS
        {
            const bf16* ga = A  + (size_t)(m0 + lrow) * K + kk + lseg;
            const bf16* gb = Wt + (size_t)(n0 + lrow) * K + kk + lseg;
            ASYNC_LD16(&As[lrow][lseg], ga, 0);
            ASYNC_LD16(&As[lrow][lseg], ga, 16);
            ASYNC_LD16(&Bs[lrow][lseg], gb, 0);
            ASYNC_LD16(&Bs[lrow][lseg], gb, 16);
        }
        ASYNC_WAIT();
#else
        {
            const uint4* src = reinterpret_cast<const uint4*>(
                A + (size_t)(m0 + lrow) * K + kk + lseg);
            uint4* dst = reinterpret_cast<uint4*>(&As[lrow][lseg]);
            dst[0] = src[0]; dst[1] = src[1];
        }
        {
            const uint4* src = reinterpret_cast<const uint4*>(
                Wt + (size_t)(n0 + lrow) * K + kk + lseg);
            uint4* dst = reinterpret_cast<uint4*>(&Bs[lrow][lseg]);
            dst[0] = src[0]; dst[1] = src[1];
        }
#endif
        __syncthreads();

        Frag16 af[2], bfg[4];
        const int ka = (lane < 16) ? 0 : 8;
        const int kb = (lane < 16) ? 0 : 8;
#pragma unroll
        for (int i = 0; i < 2; ++i) {
            const int ar = wm * 32 + i * 16 + (lane & 15);
            af[i].q[0] = *reinterpret_cast<const uint4*>(&As[ar][ka]);
            af[i].q[1] = *reinterpret_cast<const uint4*>(&As[ar][16 + ka]);
        }
#pragma unroll
        for (int j = 0; j < 4; ++j) {
            const int br = wn * 64 + j * 16 + (lane & 15);
            bfg[j].q[0] = *reinterpret_cast<const uint4*>(&Bs[br][kb]);
            bfg[j].q[1] = *reinterpret_cast<const uint4*>(&Bs[br][16 + kb]);
        }
#pragma unroll
        for (int i = 0; i < 2; ++i)
#pragma unroll
            for (int j = 0; j < 4; ++j)
                acc[i][j] = __builtin_amdgcn_wmma_f32_16x16x32_bf16(
                    false, af[i].v, false, bfg[j].v, (short)0, acc[i][j], false, false);
        __syncthreads();
    }

    const int half8 = (lane >> 4) * 8;
#pragma unroll
    for (int i = 0; i < 2; ++i) {
#pragma unroll
        for (int j = 0; j < 4; ++j) {
#pragma unroll
            for (int r = 0; r < 8; ++r) {
                const int grow = m0 + wm * 32 + i * 16 + r + half8;
                const int gcol = n0 + wn * 64 + j * 16 + (lane & 15);
                float val = acc[i][j][r];
                if (bias)    val += bias[gcol];
                if (do_gelu) val = 0.5f * val * (1.0f + erff(val * 0.70710678118f));
                if (rowvalid && !rowvalid[grow]) val = 0.0f;
                if (residual) val += residual[(size_t)grow * N + gcol];
                if (outf) outf[(size_t)grow * N + gcol] = val;
                else      outb[(size_t)grow * N + gcol] = (bf16)val;
            }
        }
    }
}

// ---------------------------------------------------------------------------
// Flash attention: grid (S/64, B*H), block 128 (4 waves x 16 q-rows).
// ---------------------------------------------------------------------------
__global__ __launch_bounds__(128)
void flash_attn_wmma(const bf16* __restrict__ q, const bf16* __restrict__ k,
                     const bf16* __restrict__ v, const int* __restrict__ mask,
                     bf16* __restrict__ ctx)
{
    constexpr int S = 1024, D = 1024, H = 16;
    __shared__ bf16 Ks[32][72];      // [key][d]   scores B-operand (contig d)
    __shared__ bf16 Vt[64][40];      // [d][key]   P*V    B-operand (contig key)
    __shared__ bf16 Ps[4][16][40];   // per-wave P relayout scratch

    const int tid   = threadIdx.x;
    const int lane  = tid & 31;
    const int wq    = tid >> 5;
    const int b     = blockIdx.y / H;
    const int h     = blockIdx.y % H;
    const int qbase = blockIdx.x * 64 + wq * 16;
    const size_t bs = (size_t)b * S;
    const int hoff  = h * 64;

    Frag16 aq[2];
    {
        const int row = qbase + (lane & 15);
        const int ka  = (lane < 16) ? 0 : 8;
        const bf16* qp = q + (bs + row) * D + hoff;
#pragma unroll
        for (int c = 0; c < 2; ++c) {
            aq[c].q[0] = *reinterpret_cast<const uint4*>(qp + c * 32 + ka);
            aq[c].q[1] = *reinterpret_cast<const uint4*>(qp + c * 32 + 16 + ka);
        }
    }

    float mrow[8], lrow[8];
    v8f accO[4] = {};
#pragma unroll
    for (int r = 0; r < 8; ++r) { mrow[r] = -3.0e38f; lrow[r] = 0.0f; }

    const int skey = tid >> 2;
    const int sd0  = (tid & 3) * 16;

    for (int kb = 0; kb < S; kb += 32) {
        // stage K rows (row-major) — async straight into LDS when available
        {
            const bf16* ksrc = k + (bs + kb + skey) * D + hoff + sd0;
#ifdef USE_ASYNC_LDS
            ASYNC_LD16(&Ks[skey][sd0], ksrc, 0);
            ASYNC_LD16(&Ks[skey][sd0], ksrc, 16);
#else
            const uint4* src = reinterpret_cast<const uint4*>(ksrc);
            uint4* dst = reinterpret_cast<uint4*>(&Ks[skey][sd0]);
            dst[0] = src[0]; dst[1] = src[1];
#endif
        }
        // stage V transposed (needs the VGPR round-trip anyway)
        {
            Frag16 u;
            const uint4* src = reinterpret_cast<const uint4*>(
                v + (bs + kb + skey) * D + hoff + sd0);
            u.q[0] = src[0]; u.q[1] = src[1];
#pragma unroll
            for (int e = 0; e < 16; ++e) Vt[sd0 + e][skey] = u.h[e];
        }
#ifdef USE_ASYNC_LDS
        ASYNC_WAIT();
#endif
        __syncthreads();

        // scores: two 16-key tiles, K-dim 64 -> 2 WMMAs each
        v8f st[2] = {};
        const int kcol = lane & 15;
        const int kbB  = (lane < 16) ? 0 : 8;
#pragma unroll
        for (int t = 0; t < 2; ++t)
#pragma unroll
            for (int c = 0; c < 2; ++c) {
                Frag16 bk;
                bk.q[0] = *reinterpret_cast<const uint4*>(&Ks[t * 16 + kcol][c * 32 + kbB]);
                bk.q[1] = *reinterpret_cast<const uint4*>(&Ks[t * 16 + kcol][c * 32 + 16 + kbB]);
                st[t] = __builtin_amdgcn_wmma_f32_16x16x32_bf16(
                    false, aq[c].v, false, bk.v, (short)0, st[t], false, false);
            }

        const int half8 = (lane >> 4) * 8;
#pragma unroll
        for (int t = 0; t < 2; ++t)
#pragma unroll
            for (int r = 0; r < 8; ++r) {
                const int qrow = qbase + r + half8;
                const int col  = kb + t * 16 + kcol;
                float sv = st[t][r] * 0.125f;
                if (mask[(bs + qrow) * S + col] == 0) sv = -1.0e9f;
                st[t][r] = sv;
            }

        // online softmax (row reductions within 16-lane halves)
        float sc[8];
#pragma unroll
        for (int r = 0; r < 8; ++r) {
            float vmax = fmaxf(st[0][r], st[1][r]);
#pragma unroll
            for (int off = 1; off < 16; off <<= 1)
                vmax = fmaxf(vmax, __shfl_xor(vmax, off, 32));
            const float mnew = fmaxf(mrow[r], vmax);
            sc[r] = __expf(mrow[r] - mnew);
            const float p0 = __expf(st[0][r] - mnew);
            const float p1 = __expf(st[1][r] - mnew);
            st[0][r] = p0; st[1][r] = p1;
            float psum = p0 + p1;
#pragma unroll
            for (int off = 1; off < 16; off <<= 1)
                psum += __shfl_xor(psum, off, 32);
            lrow[r] = lrow[r] * sc[r] + psum;
            mrow[r] = mnew;
        }
#pragma unroll
        for (int j = 0; j < 4; ++j)
#pragma unroll
            for (int r = 0; r < 8; ++r) accO[j][r] *= sc[r];

        // relayout P (C-layout -> A-operand) through per-wave LDS
#pragma unroll
        for (int t = 0; t < 2; ++t)
#pragma unroll
            for (int r = 0; r < 8; ++r)
                Ps[wq][r + half8][t * 16 + kcol] = (bf16)st[t][r];
        asm volatile("s_wait_dscnt 0x0" ::: "memory");   // in-wave LDS RAW fence

        Frag16 pf;
        {
            const int prow = lane & 15;
            const int ka   = (lane < 16) ? 0 : 8;
            pf.q[0] = *reinterpret_cast<const uint4*>(&Ps[wq][prow][ka]);
            pf.q[1] = *reinterpret_cast<const uint4*>(&Ps[wq][prow][16 + ka]);
        }
#pragma unroll
        for (int j = 0; j < 4; ++j) {
            Frag16 vf;
            const int d  = j * 16 + (lane & 15);
            const int k0 = (lane < 16) ? 0 : 8;
            vf.q[0] = *reinterpret_cast<const uint4*>(&Vt[d][k0]);
            vf.q[1] = *reinterpret_cast<const uint4*>(&Vt[d][16 + k0]);
            accO[j] = __builtin_amdgcn_wmma_f32_16x16x32_bf16(
                false, pf.v, false, vf.v, (short)0, accO[j], false, false);
        }
        __syncthreads();
    }

    const int half8 = (lane >> 4) * 8;
#pragma unroll
    for (int r = 0; r < 8; ++r) {
        const float inv = 1.0f / lrow[r];
        const int qrow = qbase + r + half8;
#pragma unroll
        for (int j = 0; j < 4; ++j)
            ctx[(bs + qrow) * D + hoff + j * 16 + (lane & 15)] =
                (bf16)(accO[j][r] * inv);
    }
}

// ---------------------------------------------------------------------------
__global__ __launch_bounds__(256)
void layernorm_to_bf16(const float* __restrict__ x, const float* __restrict__ g,
                       const float* __restrict__ bta, bf16* __restrict__ out)
{
    constexpr int D = 1024;
    __shared__ float wsum[8], wsq[8], stats[2];
    const int row = blockIdx.x;
    const int tid = threadIdx.x;
    const float4 x4 = reinterpret_cast<const float4*>(x + (size_t)row * D)[tid];
    float s  = x4.x + x4.y + x4.z + x4.w;
    float sq = x4.x * x4.x + x4.y * x4.y + x4.z * x4.z + x4.w * x4.w;
#pragma unroll
    for (int off = 1; off < 32; off <<= 1) {
        s  += __shfl_xor(s, off, 32);
        sq += __shfl_xor(sq, off, 32);
    }
    if ((tid & 31) == 0) { wsum[tid >> 5] = s; wsq[tid >> 5] = sq; }
    __syncthreads();
    if (tid == 0) {
        float ts = 0.f, tq = 0.f;
        for (int i = 0; i < 8; ++i) { ts += wsum[i]; tq += wsq[i]; }
        const float mean = ts * (1.0f / D);
        const float var  = tq * (1.0f / D) - mean * mean;
        stats[0] = mean;
        stats[1] = rsqrtf(var + 1e-5f);
    }
    __syncthreads();
    const float mean = stats[0], rstd = stats[1];
    const float4 g4 = reinterpret_cast<const float4*>(g)[tid];
    const float4 b4 = reinterpret_cast<const float4*>(bta)[tid];
    bf16* o = out + (size_t)row * D + tid * 4;
    o[0] = (bf16)((x4.x - mean) * rstd * g4.x + b4.x);
    o[1] = (bf16)((x4.y - mean) * rstd * g4.y + b4.y);
    o[2] = (bf16)((x4.z - mean) * rstd * g4.z + b4.z);
    o[3] = (bf16)((x4.w - mean) * rstd * g4.w + b4.w);
}

__global__ void cvt_f32_bf16(const float* __restrict__ in, bf16* __restrict__ out, int n)
{
    int i = blockIdx.x * blockDim.x + threadIdx.x;
    const int stride = gridDim.x * blockDim.x;
    for (; i < n; i += stride) out[i] = (bf16)in[i];
}

__global__ __launch_bounds__(256)
void row_valid_kernel(const int* __restrict__ mask, int* __restrict__ rv)
{
    const int lane = threadIdx.x & 31;
    const int row  = blockIdx.x * 8 + (threadIdx.x >> 5);
    const int* mr  = mask + (size_t)row * 1024;
    int s = 0;
    for (int i = lane; i < 1024; i += 32) s += mr[i];
#pragma unroll
    for (int off = 1; off < 32; off <<= 1) s += __shfl_xor(s, off, 32);
    if (lane == 0) rv[row] = (s > 0) ? 1 : 0;
}

// ---------------------------------------------------------------------------
extern "C" void kernel_launch(void* const* d_in, const int* in_sizes, int n_in,
                              void* d_out, int out_size, void* d_ws, size_t ws_size,
                              hipStream_t stream)
{
    constexpr int B = 4, S = 1024, D = 1024, H = 16, F = 4096, M = B * S;

    const float* x     = (const float*)d_in[0];
    const float* Wq    = (const float*)d_in[1];
    const float* Wk    = (const float*)d_in[2];
    const float* Wv    = (const float*)d_in[3];
    const float* Wo    = (const float*)d_in[4];
    const float* w1    = (const float*)d_in[5];
    const float* b1    = (const float*)d_in[6];
    const float* w2    = (const float*)d_in[7];
    const float* b2    = (const float*)d_in[8];
    const float* ln1_g = (const float*)d_in[9];
    const float* ln1_b = (const float*)d_in[10];
    const float* ln2_g = (const float*)d_in[11];
    const float* ln2_b = (const float*)d_in[12];
    const int*   mask  = (const int*)d_in[13];
    float* out = (float*)d_out;

    char* ws = (char*)d_ws;
    size_t off = 0;
    auto carve = [&](size_t bytes) {
        void* p = ws + off;
        off += (bytes + 255) & ~(size_t)255;
        return p;
    };
    bf16* hbf  = (bf16*)carve((size_t)M * D * 2);
    bf16* wqb  = (bf16*)carve((size_t)D * D * 2);
    bf16* wkb  = (bf16*)carve((size_t)D * D * 2);
    bf16* wvb  = (bf16*)carve((size_t)D * D * 2);
    bf16* wob  = (bf16*)carve((size_t)D * D * 2);
    bf16* w1b  = (bf16*)carve((size_t)F * D * 2);
    bf16* w2b  = (bf16*)carve((size_t)D * F * 2);
    bf16* qb   = (bf16*)carve((size_t)M * D * 2);
    bf16* kb   = (bf16*)carve((size_t)M * D * 2);
    bf16* vb   = (bf16*)carve((size_t)M * D * 2);
    bf16* ctxb = (bf16*)carve((size_t)M * D * 2);
    float* x1  = (float*)carve((size_t)M * D * 4);
    bf16* h2b  = (bf16*)carve((size_t)M * D * 2);
    bf16* a1b  = (bf16*)carve((size_t)M * F * 2);
    int*  rvp  = (int*)carve((size_t)M * 4);

    cvt_f32_bf16<<<4096, 256, 0, stream>>>(Wq, wqb, D * D);
    cvt_f32_bf16<<<4096, 256, 0, stream>>>(Wk, wkb, D * D);
    cvt_f32_bf16<<<4096, 256, 0, stream>>>(Wv, wvb, D * D);
    cvt_f32_bf16<<<4096, 256, 0, stream>>>(Wo, wob, D * D);
    cvt_f32_bf16<<<8192, 256, 0, stream>>>(w1, w1b, F * D);
    cvt_f32_bf16<<<8192, 256, 0, stream>>>(w2, w2b, D * F);

    layernorm_to_bf16<<<M, 256, 0, stream>>>(x, ln1_g, ln1_b, hbf);

    dim3 gqkv(D / 128, M / 128);
    gemm_bf16_wmma<<<gqkv, 256, 0, stream>>>(hbf, wqb, nullptr, nullptr, nullptr,
                                             qb, nullptr, M, D, D, 0);
    gemm_bf16_wmma<<<gqkv, 256, 0, stream>>>(hbf, wkb, nullptr, nullptr, nullptr,
                                             kb, nullptr, M, D, D, 0);
    gemm_bf16_wmma<<<gqkv, 256, 0, stream>>>(hbf, wvb, nullptr, nullptr, nullptr,
                                             vb, nullptr, M, D, D, 0);

    flash_attn_wmma<<<dim3(S / 64, B * H), 128, 0, stream>>>(qb, kb, vb, mask, ctxb);

    gemm_bf16_wmma<<<gqkv, 256, 0, stream>>>(ctxb, wob, nullptr, x, nullptr,
                                             nullptr, x1, M, D, D, 0);

    layernorm_to_bf16<<<M, 256, 0, stream>>>(x1, ln2_g, ln2_b, h2b);
    row_valid_kernel<<<M / 8, 256, 0, stream>>>(mask, rvp);

    gemm_bf16_wmma<<<dim3(F / 128, M / 128), 256, 0, stream>>>(
        h2b, w1b, b1, nullptr, nullptr, a1b, nullptr, M, F, D, 1);

    gemm_bf16_wmma<<<dim3(D / 128, M / 128), 256, 0, stream>>>(
        a1b, w2b, b2, x1, rvp, nullptr, out, M, D, F, 0);
}
